// FlowMatchingPolicy_64716567216769
// MI455X (gfx1250) — matrix operand order repaired and checked by reading, not compiled, not run on hardware
//
#include <hip/hip_runtime.h>
#include <cstdint>

// ---------------- problem constants ----------------
#define B_    2048
#define SD_   256
#define AD_   7
#define T_    10
#define E_    256
#define NL_   3
#define DI_   512
#define DS_   64
#define DC_   4
#define DR_   16
#define DINT_ 256
#define L_    12          // 1 + 1 + T
#define XPN_  144         // DR + 2*DS

typedef unsigned short u16;
typedef unsigned int   u32;

// ---------------- WMMA types (CDNA5 wave32) ----------------
typedef __attribute__((ext_vector_type(16))) __bf16 v16bf;
typedef __attribute__((ext_vector_type(8)))  float  v8f;

union Frag { v16bf b; uint4 q[2]; };   // 32 bytes: two b128 chunks

__device__ __forceinline__ u16 f2bf(float f) {
  u32 x = __builtin_bit_cast(u32, f);
  x += 0x7FFFu + ((x >> 16) & 1u);     // round-to-nearest-even
  return (u16)(x >> 16);
}
__device__ __forceinline__ float bf2f(u16 u) {
  return __builtin_bit_cast(float, (u32)u << 16);
}
__device__ __forceinline__ float siluf_(float x) { return x / (1.f + __expf(-x)); }

// =====================================================================
// bf16 WMMA GEMM:  C = EPI(A[M,K] @ W[N,K]^T + bias + add2) (+resid)
// A, W are bf16 (pre-converted). A/C rows remapped via (group,stride,off).
// EPI: 0=none 1=silu 2=relu.  OUTBF: store bf16 instead of fp32.
// Wave computes 32x64 tile (2x4 16x16x32 fragments); 4 waves (2x2) give a
// 64x128 block tile. Inner loop: 12x global_load_b128 + 8x v_wmma.
// =====================================================================
template<int EPI, bool HAS_BIAS, bool HAS_ADD2, bool HAS_RES, bool OUTBF>
__global__ __launch_bounds__(128)
void gemm_bf16_wmma(const u16* __restrict__ A, int lda,
                    int aGroup, int aStr, int aOff,
                    const u16* __restrict__ W,        // [N,K] row-major bf16
                    const float* __restrict__ bias,
                    const float* __restrict__ add2,
                    const float* __restrict__ resid,
                    void* __restrict__ Cv, int ldc,
                    int cGroup, int cStr, int cOff,
                    int M, int N, int K)
{
  const int lane = threadIdx.x & 31;
  const int wv   = threadIdx.x >> 5;
  const int wR   = wv >> 1, wC = wv & 1;
  const int mBase = blockIdx.y * 64 + wR * 32;
  const int nBase = blockIdx.x * 128 + wC * 64;
  const int lm = lane & 15;
  const int hf = lane >> 4;

  v8f acc[2][4] = {};

  long aRowOffs[2];
  #pragma unroll
  for (int i = 0; i < 2; ++i) {
    int m = mBase + i * 16 + lm;
    if (m >= M) m = M - 1;             // clamp loads; stores guarded below
    long r = (long)(m / aGroup) * aStr + aOff + (m % aGroup);
    aRowOffs[i] = r * (long)lda;
  }
  int nIdx[4]; bool nOK[4];
  #pragma unroll
  for (int j = 0; j < 4; ++j) {
    int n = nBase + j * 16 + lm;
    nOK[j] = (n < N);
    nIdx[j] = nOK[j] ? n : 0;
  }

  for (int kb = 0; kb < K; kb += 32) {
    Frag a[2], b[4];
    #pragma unroll
    for (int i = 0; i < 2; ++i) {
      // A 16x32 bf16 layout: lane half hf -> K = hf*8+{0..7} and 16+hf*8+{0..7}
      const u16* ap = A + aRowOffs[i] + kb + hf * 8;
      a[i].q[0] = *(const uint4*)(ap);
      a[i].q[1] = *(const uint4*)(ap + 16);
      __builtin_prefetch(ap + 32, 0, 1);   // global_prefetch_b8 (safe if OOB)
    }
    #pragma unroll
    for (int j = 0; j < 4; ++j) {
      // B 32x16 bf16 layout: lane half hf -> K = hf*8+{0..7} and 16+hf*8+{0..7}
      const u16* wp = W + (long)nIdx[j] * K + kb + hf * 8;
      b[j].q[0] = *(const uint4*)(wp);
      b[j].q[1] = *(const uint4*)(wp + 16);
      __builtin_prefetch(wp + 32, 0, 1);
    }
    #pragma unroll
    for (int i = 0; i < 2; ++i)
      #pragma unroll
      for (int j = 0; j < 4; ++j)
        acc[i][j] = __builtin_amdgcn_wmma_f32_16x16x32_bf16(
            false, a[i].b, false, b[j].b, (short)0, acc[i][j], false, false);
  }

  // epilogue: C/D layout -> lane n = lm, v8f element r is row hf*8 + r
  #pragma unroll
  for (int j = 0; j < 4; ++j) {
    if (!nOK[j]) continue;
    const int n = nIdx[j];
    const float bval = (HAS_BIAS ? bias[n] : 0.f) + (HAS_ADD2 ? add2[n] : 0.f);
    #pragma unroll
    for (int i = 0; i < 2; ++i) {
      #pragma unroll
      for (int r = 0; r < 8; ++r) {
        int m = mBase + i * 16 + hf * 8 + r;
        if (m >= M) continue;
        float v = acc[i][j][r] + bval;
        if (EPI == 1) v = siluf_(v);
        if (EPI == 2) v = fmaxf(v, 0.f);
        long crow = (long)(m / cGroup) * cStr + cOff + (m % cGroup);
        long off = crow * (long)ldc + n;
        if (HAS_RES) v += resid[off];
        if (OUTBF) ((u16*)Cv)[off] = f2bf(v);
        else       ((float*)Cv)[off] = v;
      }
    }
  }
}

// ---------------- fp32 -> bf16 bulk converter (weights, z_next) ----------
__global__ void cvt_bf16_kernel(const float* __restrict__ in, u16* __restrict__ out, long n) {
  long i = (long)blockIdx.x * blockDim.x + threadIdx.x;
  if (i < n) out[i] = f2bf(in[i]);
}

// ---------------- sinusoidal time features (bf16 out) ----------------
__global__ void timefeat_kernel(const float* __restrict__ sigma, u16* __restrict__ emb) {
  long i = (long)blockIdx.x * blockDim.x + threadIdx.x;   // B*E
  int j = (int)(i % E_); long b = i / E_;
  int h = (j < 128) ? j : j - 128;
  float freq = __expf(-logf(10000.f) * (float)h / 128.f);
  float arg = sigma[b] * freq;
  emb[i] = f2bf((j < 128) ? __sinf(arg) : __cosf(arg));
}

// ---------------- action token embedding (K=7, VALU, fp32 out into x) ----
__global__ void act_embed_kernel(const float* __restrict__ na, const float* __restrict__ aw,
                                 const float* __restrict__ ab, const float* __restrict__ pe,
                                 float* __restrict__ x) {
  long i = (long)blockIdx.x * blockDim.x + threadIdx.x;   // B*T*E
  int e = (int)(i % E_); long bt = i / E_;
  int t = (int)(bt % T_); long b = bt / T_;
  float acc = ab[e];
  #pragma unroll
  for (int a = 0; a < AD_; ++a) acc += na[(b * T_ + t) * AD_ + a] * aw[e * AD_ + a];
  acc += pe[(2 + t) * E_ + e];
  x[(b * L_ + 2 + t) * (long)E_ + e] = acc;
}

// ---------------- RMSNorm: fp32 in, bf16 out; one wave per row ----------
__global__ __launch_bounds__(256)
void rmsnorm_kernel(const float* __restrict__ x, const float* __restrict__ w,
                    u16* __restrict__ out, long rows) {
  int lane = threadIdx.x & 31;
  long row = (long)blockIdx.x * (blockDim.x >> 5) + (threadIdx.x >> 5);
  if (row >= rows) return;
  const float* xr = x + row * E_;
  float v[8]; float ss = 0.f;
  #pragma unroll
  for (int i = 0; i < 8; ++i) { v[i] = xr[lane + 32 * i]; ss += v[i] * v[i]; }
  #pragma unroll
  for (int m = 16; m > 0; m >>= 1) ss += __shfl_xor(ss, m, 32);
  float inv = rsqrtf(ss * (1.f / E_) + 1e-5f);
  #pragma unroll
  for (int i = 0; i < 8; ++i) { int c = lane + 32 * i; out[row * E_ + c] = f2bf(v[i] * inv * w[c]); }
}

// ---------------- causal depthwise conv (DC=4) + SiLU; bf16 out ---------
__global__ void conv_kernel(const float* __restrict__ xz, const float* __restrict__ cw,
                            const float* __restrict__ cb, u16* __restrict__ xc) {
  long i = (long)blockIdx.x * blockDim.x + threadIdx.x;   // B*L*DI
  int d = (int)(i % DI_); long bl = i / DI_;
  int l = (int)(bl % L_); long b = bl / L_;
  float acc = cb[d];
  #pragma unroll
  for (int k = 0; k < DC_; ++k) {
    int ll = l + k - (DC_ - 1);
    if (ll >= 0) acc += xz[(b * L_ + ll) * (long)(2 * DI_) + d] * cw[d * DC_ + k];
  }
  xc[i] = f2bf(siluf_(acc));
}

// ---------------- dt = softplus(dbl[:, :16] @ dt_w^T + dt_b); fp32 ------
__global__ __launch_bounds__(256)
void dt_softplus_kernel(const float* __restrict__ dbl, const float* __restrict__ dtw,
                        const float* __restrict__ dtb, float* __restrict__ dt) {
  __shared__ float s[DR_];
  long row = blockIdx.x;                                  // b*L + l
  if (threadIdx.x < DR_) s[threadIdx.x] = dbl[row * XPN_ + threadIdx.x];
  __syncthreads();
  for (int d = threadIdx.x; d < DI_; d += blockDim.x) {
    float acc = dtb[d];
    #pragma unroll
    for (int r = 0; r < DR_; ++r) acc += s[r] * dtw[d * DR_ + r];
    dt[row * DI_ + d] = (acc > 20.f) ? acc : log1pf(__expf(acc));
  }
}

// ---------------- selective scan: thread per (b,d), h[64] in registers --
// reads dt (fp32), xc (bf16), dbl (fp32), z from xz (fp32); writes y bf16
__global__ __launch_bounds__(256)
void scan_kernel(const u16* __restrict__ xc, const float* __restrict__ dbl,
                 const float* __restrict__ xz, const float* __restrict__ Alog,
                 const float* __restrict__ Dp, const float* __restrict__ dt,
                 u16* __restrict__ ybf) {
  long i = (long)blockIdx.x * blockDim.x + threadIdx.x;   // B*DI
  int d = (int)(i % DI_); long b = i / DI_;
  float a[DS_], h[DS_];
  #pragma unroll
  for (int n = 0; n < DS_; ++n) { a[n] = -__expf(Alog[d * DS_ + n]); h[n] = 0.f; }
  float Dv = Dp[d];
  for (int t = 0; t < L_; ++t) {
    long row = b * L_ + t;
    float dtv = dt[row * DI_ + d];
    float xv  = bf2f(xc[row * DI_ + d]);
    const float* Bc = dbl + row * XPN_ + DR_;
    const float* Cc = Bc + DS_;
    float dx = dtv * xv, y = 0.f;
    #pragma unroll
    for (int n = 0; n < DS_; ++n) {
      h[n] = __expf(dtv * a[n]) * h[n] + dx * Bc[n];
      y += h[n] * Cc[n];
    }
    float zv = xz[row * (long)(2 * DI_) + DI_ + d];
    ybf[row * DI_ + d] = f2bf((y + Dv * xv) * siluf_(zv));
  }
}

// ---------------- gated-MLP elementwise: h = y * silu(gate); bf16 out ---
__global__ void glu_kernel(const float* __restrict__ g, u16* __restrict__ h) {
  long i = (long)blockIdx.x * blockDim.x + threadIdx.x;   // B*L*DINT
  long row = i / DINT_; int j = (int)(i % DINT_);
  float y = g[row * (2 * DINT_) + j];
  float gate = g[row * (2 * DINT_) + DINT_ + j];
  h[row * DINT_ + j] = f2bf(y * siluf_(gate));
}

// ---------------- final head: (B*T,256) -> 7 ----------------
__global__ void head_kernel(const float* __restrict__ hdn, const float* __restrict__ w2,
                            const float* __restrict__ b2, float* __restrict__ out, long n) {
  long i = (long)blockIdx.x * blockDim.x + threadIdx.x;   // B*T*AD
  if (i >= n) return;
  int a = (int)(i % AD_); long bt = i / AD_;
  const float* hr = hdn + bt * E_;
  const float* wr = w2 + a * E_;
  float acc = b2[a];
  for (int e = 0; e < E_; ++e) acc += hr[e] * wr[e];
  out[i] = acc;
}

// =====================================================================
extern "C" void kernel_launch(void* const* d_in, const int* in_sizes, int n_in,
                              void* d_out, int out_size, void* d_ws, size_t ws_size,
                              hipStream_t stream) {
  (void)in_sizes; (void)n_in; (void)out_size; (void)ws_size;
  const float* z_next     = (const float*)d_in[0];
  const float* noisy      = (const float*)d_in[1];
  const float* sigma      = (const float*)d_in[2];
  const float* pos_emb    = (const float*)d_in[3];
  const float* state_w    = (const float*)d_in[4];
  const float* state_b    = (const float*)d_in[5];
  const float* act_w      = (const float*)d_in[6];
  const float* act_b      = (const float*)d_in[7];
  const float* t_w1       = (const float*)d_in[8];
  const float* t_b1       = (const float*)d_in[9];
  const float* t_w2       = (const float*)d_in[10];
  const float* t_b2       = (const float*)d_in[11];
  const float* norm_w     = (const float*)d_in[12];
  const float* in_proj_w  = (const float*)d_in[13];
  const float* conv_w     = (const float*)d_in[14];
  const float* conv_b     = (const float*)d_in[15];
  const float* x_proj_w   = (const float*)d_in[16];
  const float* dt_w       = (const float*)d_in[17];
  const float* dt_b       = (const float*)d_in[18];
  const float* A_log      = (const float*)d_in[19];
  const float* Dp         = (const float*)d_in[20];
  const float* out_proj_w = (const float*)d_in[21];
  const float* norm2_w    = (const float*)d_in[22];
  const float* fc1_w      = (const float*)d_in[23];
  const float* fc2_w      = (const float*)d_in[24];
  const float* norm_f_w   = (const float*)d_in[25];
  const float* pred_w1    = (const float*)d_in[26];
  const float* pred_b1    = (const float*)d_in[27];
  const float* pred_w2    = (const float*)d_in[28];
  const float* pred_b2    = (const float*)d_in[29];
  float* out = (float*)d_out;

  // ---------------- workspace carving (256B aligned) ----------------
  char* base = (char*)d_ws;
  size_t woff = 0;
  auto alloc = [&](size_t bytes) -> void* {
    void* p = base + woff;
    woff = (woff + bytes + 255) & ~(size_t)255;
    return p;
  };
  const long BLE = (long)B_ * L_ * E_;
  float* x    = (float*)alloc(BLE * 4);                    // residual stream
  float* xz   = (float*)alloc((long)B_ * L_ * 2 * DI_ * 4);
  float* dbl  = (float*)alloc((long)B_ * L_ * XPN_ * 4);
  float* dt   = (float*)alloc((long)B_ * L_ * DI_ * 4);
  u16*   xn   = (u16*)  alloc(BLE * 2);                    // rmsnorm out (bf16)
  u16*   xc   = (u16*)  alloc((long)B_ * L_ * DI_ * 2);    // conv out (bf16)
  u16*   ybf  = (u16*)  alloc((long)B_ * L_ * DI_ * 2);    // scan out (bf16)
  // bf16 weight cache
  u16* wb_tw1   = (u16*)alloc((long)E_ * E_ * 2);
  u16* wb_tw2   = (u16*)alloc((long)E_ * E_ * 2);
  u16* wb_state = (u16*)alloc((long)E_ * SD_ * 2);
  u16* wb_inp   = (u16*)alloc((long)NL_ * 2 * DI_ * E_ * 2);
  u16* wb_xp    = (u16*)alloc((long)NL_ * XPN_ * DI_ * 2);
  u16* wb_outp  = (u16*)alloc((long)NL_ * E_ * DI_ * 2);
  u16* wb_fc1   = (u16*)alloc((long)NL_ * 2 * DINT_ * E_ * 2);
  u16* wb_fc2   = (u16*)alloc((long)NL_ * E_ * DINT_ * 2);
  u16* wb_pred1 = (u16*)alloc((long)E_ * E_ * 2);
  // reused regions
  u16*   hml = ybf;            // MLP hidden (bf16), after y consumed
  float* hdn = (float*)xc;     // head hidden (fp32), after layers
  u16*   emb = (u16*)dbl;      // time features (bf16), pre-layers
  u16*   t1  = emb + (long)B_ * E_;
  u16*   zbf = t1 + (long)B_ * E_;     // z_next in bf16

  const int IDG = 1 << 30;     // identity row mapping
  const int MR  = B_ * L_;     // 24576

  // ---- weight / input conversion to bf16 (few MB; once per launch) ----
  auto cvt = [&](const float* s, u16* d2, long n) {
    cvt_bf16_kernel<<<(n + 255) / 256, 256, 0, stream>>>(s, d2, n);
  };
  cvt(t_w1, wb_tw1, (long)E_ * E_);
  cvt(t_w2, wb_tw2, (long)E_ * E_);
  cvt(state_w, wb_state, (long)E_ * SD_);
  cvt(in_proj_w, wb_inp, (long)NL_ * 2 * DI_ * E_);
  cvt(x_proj_w, wb_xp, (long)NL_ * XPN_ * DI_);
  cvt(out_proj_w, wb_outp, (long)NL_ * E_ * DI_);
  cvt(fc1_w, wb_fc1, (long)NL_ * 2 * DINT_ * E_);
  cvt(fc2_w, wb_fc2, (long)NL_ * E_ * DINT_);
  cvt(pred_w1, wb_pred1, (long)E_ * E_);
  cvt(z_next, zbf, (long)B_ * SD_);

  // ---- embeddings ----
  timefeat_kernel<<<(B_ * E_) / 256, 256, 0, stream>>>(sigma, emb);
  gemm_bf16_wmma<1, true, false, false, true><<<dim3((E_ + 127) / 128, B_ / 64), 128, 0, stream>>>(
      emb, E_, IDG, 0, 0, wb_tw1, t_b1, nullptr, nullptr, t1, E_, IDG, 0, 0, B_, E_, E_);
  gemm_bf16_wmma<0, true, true, false, false><<<dim3((E_ + 127) / 128, B_ / 64), 128, 0, stream>>>(
      t1, E_, IDG, 0, 0, wb_tw2, t_b2, pos_emb + 0 * E_, nullptr, x, E_, 1, L_, 0, B_, E_, E_);
  gemm_bf16_wmma<0, true, true, false, false><<<dim3((E_ + 127) / 128, B_ / 64), 128, 0, stream>>>(
      zbf, SD_, IDG, 0, 0, wb_state, state_b, pos_emb + 1 * E_, nullptr, x, E_, 1, L_, 1, B_, E_, SD_);
  act_embed_kernel<<<(B_ * T_ * E_) / 256, 256, 0, stream>>>(noisy, act_w, act_b, pos_emb, x);

  // ---- Mamba layers ----
  for (int l = 0; l < NL_; ++l) {
    rmsnorm_kernel<<<(MR + 7) / 8, 256, 0, stream>>>(x, norm_w + l * E_, xn, (long)MR);
    gemm_bf16_wmma<0, false, false, false, false><<<dim3((2 * DI_) / 128, MR / 64), 128, 0, stream>>>(
        xn, E_, IDG, 0, 0, wb_inp + (long)l * 2 * DI_ * E_, nullptr, nullptr, nullptr,
        xz, 2 * DI_, IDG, 0, 0, MR, 2 * DI_, E_);
    conv_kernel<<<(MR * DI_) / 256, 256, 0, stream>>>(xz, conv_w + l * DI_ * DC_, conv_b + l * DI_, xc);
    gemm_bf16_wmma<0, false, false, false, false><<<dim3((XPN_ + 127) / 128, MR / 64), 128, 0, stream>>>(
        xc, DI_, IDG, 0, 0, wb_xp + (long)l * XPN_ * DI_, nullptr, nullptr, nullptr,
        dbl, XPN_, IDG, 0, 0, MR, XPN_, DI_);
    dt_softplus_kernel<<<MR, 256, 0, stream>>>(dbl, dt_w + l * DI_ * DR_, dt_b + l * DI_, dt);
    scan_kernel<<<(B_ * DI_) / 256, 256, 0, stream>>>(xc, dbl, xz,
        A_log + (long)l * DI_ * DS_, Dp + l * DI_, dt, ybf);
    gemm_bf16_wmma<0, false, false, true, false><<<dim3((E_ + 127) / 128, MR / 64), 128, 0, stream>>>(
        ybf, DI_, IDG, 0, 0, wb_outp + (long)l * E_ * DI_, nullptr, nullptr, x,
        x, E_, IDG, 0, 0, MR, E_, DI_);
    // gated MLP
    rmsnorm_kernel<<<(MR + 7) / 8, 256, 0, stream>>>(x, norm2_w + l * E_, xn, (long)MR);
    gemm_bf16_wmma<0, false, false, false, false><<<dim3((2 * DINT_) / 128, MR / 64), 128, 0, stream>>>(
        xn, E_, IDG, 0, 0, wb_fc1 + (long)l * 2 * DINT_ * E_, nullptr, nullptr, nullptr,
        xz, 2 * DINT_, IDG, 0, 0, MR, 2 * DINT_, E_);
    glu_kernel<<<(MR * DINT_) / 256, 256, 0, stream>>>(xz, hml);
    gemm_bf16_wmma<0, false, false, true, false><<<dim3((E_ + 127) / 128, MR / 64), 128, 0, stream>>>(
        hml, DINT_, IDG, 0, 0, wb_fc2 + (long)l * E_ * DINT_, nullptr, nullptr, x,
        x, E_, IDG, 0, 0, MR, E_, DINT_);
  }

  // ---- head ----
  rmsnorm_kernel<<<(MR + 7) / 8, 256, 0, stream>>>(x, norm_f_w, xn, (long)MR);
  gemm_bf16_wmma<2, true, false, false, false><<<dim3((E_ + 127) / 128, (B_ * T_) / 64), 128, 0, stream>>>(
      xn, E_, T_, L_, 2, wb_pred1, pred_b1, nullptr, nullptr,
      hdn, E_, IDG, 0, 0, B_ * T_, E_, E_);
  head_kernel<<<((long)B_ * T_ * AD_ + 255) / 256, 256, 0, stream>>>(
      hdn, pred_w2, pred_b2, out, (long)B_ * T_ * AD_);
}
